// GATMamba_27951647162847
// MI455X (gfx1250) — compile-verified, matrix operand order checked by reading, not run on hardware
//
#include <hip/hip_runtime.h>

typedef __attribute__((ext_vector_type(16))) __bf16 v16bf;
typedef __attribute__((ext_vector_type(8)))  __bf16 v8bf;
typedef __attribute__((ext_vector_type(8)))  float  v8f;
typedef __attribute__((ext_vector_type(2)))  float  f2v;

#define CDIV(a,b) (((a)+(b)-1)/(b))

constexpr int NN     = 50000;   // nodes
constexpr int NE     = 800000;  // edges
constexpr int NG     = 50;      // graphs
constexpr int LG     = 1000;    // nodes per graph
constexpr int CH     = 80;
constexpr int EEMB   = 16;
constexpr int DSTATE = 16;
constexpr int DTR    = 5;
constexpr int PROJW  = DTR + 2 * DSTATE;  // 37
constexpr int DUNI   = 1024;

__device__ __forceinline__ float sigmoidf_(float x) { return 1.f / (1.f + expf(-x)); }
__device__ __forceinline__ float siluf_(float x)    { return x * sigmoidf_(x); }
__device__ __forceinline__ float leakyf_(float x)   { return x > 0.f ? x : 0.2f * x; }
__device__ __forceinline__ float bnf_(float x, float w, float b, float m, float v) {
  return (x - m) * rsqrtf(v + 1e-5f) * w + b;
}

__device__ __forceinline__ void atomicMaxF(float* addr, float val) {
  unsigned int* ua = (unsigned int*)addr;
  unsigned int cur = *ua;
  while (__uint_as_float(cur) < val) {
    unsigned int prev = atomicCAS(ua, cur, __float_as_uint(val));
    if (prev == cur) break;
    cur = prev;
  }
}

// ---------------------------------------------------------------------------
// bf16 WMMA GEMM: C[M,N] = A[M,K] * B[K,N] (+ bias[N]); fp32 in/out.
// Block = 256 thr = 8 waves. Block tile 64(M) x 64(N), K-step 32.
// Wave w: row tile rt = w&3; col tiles (w>>2) and (w>>2)+2 share one A frag
// -> 2x v_wmma_f32_16x16x32_bf16 per wave per k-step.
// LDS: A row-major [64][40], B transposed [col][k] = [64][40]; 40-elt rows
// give 16B-aligned, bank-conflict-free 8xbf16 runs -> ds_load_b128 fragments.
// ---------------------------------------------------------------------------
#define TK 32
__global__ __launch_bounds__(256)
void gemm_bf16_wmma(const float* __restrict__ A, int lda,
                    const float* __restrict__ B, int ldb,
                    const float* __restrict__ bias,
                    float* __restrict__ C, int ldc,
                    int M, int N, int K) {
  __shared__ __align__(16) __bf16 As[64][40];
  __shared__ __align__(16) __bf16 Bst[64][40];   // [column][k]

  const int tid  = threadIdx.x;
  const int wave = tid >> 5;
  const int lane = tid & 31;
  const int rt   = wave & 3;        // row tile (0..3)
  const int cth  = wave >> 2;       // col tile pair selector (0..1)
  const int half = lane >> 4;       // K-half selector
  const int lrow = lane & 15;
  const int blockRow = blockIdx.x * 64;
  const int blockCol = blockIdx.y * 64;

  // staging assignments: A: row r = tid>>2, seg = tid&3 (8 k each)
  //                      B: col c = tid>>2, seg = tid&3 (8 k each)
  const int sr  = tid >> 2;
  const int seg = tid & 3;

  v8f acc0 = {}, acc1 = {};

  for (int k0 = 0; k0 < K; k0 += TK) {
    // ---- stage A tile 64x32 ----
    {
      const int gr  = blockRow + sr;
      const int gk0 = k0 + seg * 8;
      float vals[8];
      if (gr < M && gk0 + 8 <= K) {
        const float* ap = A + (long)gr * lda + gk0;
#pragma unroll
        for (int i = 0; i < 4; ++i) {
          f2v t = *(const f2v*)(ap + 2 * i);
          vals[2 * i] = t.x; vals[2 * i + 1] = t.y;
        }
      } else {
#pragma unroll
        for (int i = 0; i < 8; ++i) {
          int gk = gk0 + i;
          vals[i] = (gr < M && gk < K) ? A[(long)gr * lda + gk] : 0.f;
        }
      }
      v8bf pk;
#pragma unroll
      for (int i = 0; i < 8; ++i) pk[i] = (__bf16)vals[i];
      *(v8bf*)&As[sr][seg * 8] = pk;
    }
    // ---- stage B tile 32x64, transposed into Bst[col][k] ----
    {
      const int gc  = blockCol + sr;
      const int gk0 = k0 + seg * 8;
      float vals[8];
      if (gc < N && gk0 + 8 <= K) {
#pragma unroll
        for (int i = 0; i < 8; ++i) vals[i] = B[(long)(gk0 + i) * ldb + gc];
      } else {
#pragma unroll
        for (int i = 0; i < 8; ++i) {
          int gk = gk0 + i;
          vals[i] = (gc < N && gk < K) ? B[(long)gk * ldb + gc] : 0.f;
        }
      }
      v8bf pk;
#pragma unroll
      for (int i = 0; i < 8; ++i) pk[i] = (__bf16)vals[i];
      *(v8bf*)&Bst[sr][seg * 8] = pk;
    }
    // prefetch next A tile into cache (speculative)
    if (k0 + TK < K) {
      int pr = blockRow + (tid >> 2);
      int pk = k0 + TK + ((tid & 3) << 3);
      if (pr < M && pk < K) __builtin_prefetch(&A[(long)pr * lda + pk], 0, 0);
    }
    __syncthreads();

    // ---- fragments via 16B LDS loads ----
    const int arow  = rt * 16 + lrow;
    const int bcol0 = cth * 16 + lrow;
    v8bf a_lo = *(const v8bf*)&As[arow][half * 8];
    v8bf a_hi = *(const v8bf*)&As[arow][16 + half * 8];
    v16bf a = __builtin_shufflevector(a_lo, a_hi, 0, 1, 2, 3, 4, 5, 6, 7,
                                      8, 9, 10, 11, 12, 13, 14, 15);
    v8bf b0_lo = *(const v8bf*)&Bst[bcol0][half * 8];
    v8bf b0_hi = *(const v8bf*)&Bst[bcol0][16 + half * 8];
    v16bf b0 = __builtin_shufflevector(b0_lo, b0_hi, 0, 1, 2, 3, 4, 5, 6, 7,
                                       8, 9, 10, 11, 12, 13, 14, 15);
    v8bf b1_lo = *(const v8bf*)&Bst[bcol0 + 32][half * 8];
    v8bf b1_hi = *(const v8bf*)&Bst[bcol0 + 32][16 + half * 8];
    v16bf b1 = __builtin_shufflevector(b1_lo, b1_hi, 0, 1, 2, 3, 4, 5, 6, 7,
                                       8, 9, 10, 11, 12, 13, 14, 15);

    acc0 = __builtin_amdgcn_wmma_f32_16x16x32_bf16(false, a, false, b0,
                                                   (short)0, acc0, false, false);
    acc1 = __builtin_amdgcn_wmma_f32_16x16x32_bf16(false, a, false, b1,
                                                   (short)0, acc1, false, false);
    __syncthreads();
  }

#pragma unroll
  for (int r = 0; r < 8; ++r) {
    int row  = blockRow + rt * 16 + ((lane < 16) ? r : r + 8);
    int col0 = blockCol + cth * 16 + lrow;
    int col1 = col0 + 32;
    if (row < M) {
      if (col0 < N) {
        float v = acc0[r];
        if (bias) v += bias[col0];
        C[(long)row * ldc + col0] = v;
      }
      if (col1 < N) {
        float v = acc1[r];
        if (bias) v += bias[col1];
        C[(long)row * ldc + col1] = v;
      }
    }
  }
}

// ---------------------------------------------------------------------------
// small elementwise kernels
// ---------------------------------------------------------------------------
__global__ void zero_kernel(float* p, long n) {
  long i = (long)blockIdx.x * blockDim.x + threadIdx.x;
  if (i < n) p[i] = 0.f;
}

__global__ void relu_kernel(float* p, long n) {
  long i = (long)blockIdx.x * blockDim.x + threadIdx.x;
  if (i < n) p[i] = fmaxf(p[i], 0.f);
}

// sinusoidal PE into h[:, 64:80]
__global__ void pe_kernel(const float* __restrict__ x, float* __restrict__ h) {
  int n = blockIdx.x * blockDim.x + threadIdx.x;
  if (n >= NN) return;
  float p0 = x[(long)n * (DUNI + 2) + 0];
  float p1 = x[(long)n * (DUNI + 2) + 1];
#pragma unroll
  for (int i = 0; i < 4; ++i) {
    float den = powf(10000.f, (float)i * 0.25f);  // 10000^(2i/8)
    float a0 = p0 / den, a1 = p1 / den;
    h[(long)n * CH + 64 + 2 * i]     = sinf(a0);
    h[(long)n * CH + 64 + 2 * i + 1] = cosf(a0);
    h[(long)n * CH + 72 + 2 * i]     = sinf(a1);
    h[(long)n * CH + 72 + 2 * i + 1] = cosf(a1);
  }
}

// weae[j] = sum_c gat_We[j,c] * gat_ae[c]   (fold (ea@We)@ae -> ea@(We@ae))
__global__ void weae_kernel(const float* __restrict__ We,
                            const float* __restrict__ ae,
                            float* __restrict__ weae) {
  int j = threadIdx.x;
  if (j >= EEMB) return;
  float s = 0.f;
  for (int c = 0; c < CH; ++c) s += We[j * CH + c] * ae[c];
  weae[j] = s;
}

// per-edge: eattr, a_e, node sums for self-loop fill_value='mean'
__global__ void edge_attr_kernel(const int* __restrict__ ea_cat,
                                 const float* __restrict__ ea_cont,
                                 const float* __restrict__ W_edge,
                                 const float* __restrict__ b_edge,
                                 const float* __restrict__ edge_emb,
                                 const float* __restrict__ weae,
                                 const int* __restrict__ dste,
                                 float* __restrict__ eattr,
                                 float* __restrict__ a_e,
                                 float* __restrict__ nodesum,
                                 float* __restrict__ cnt) {
  int e = blockIdx.x * blockDim.x + threadIdx.x;
  if (e >= NE) return;
  float c0 = ea_cont[(long)e * 2 + 0];
  float c1 = ea_cont[(long)e * 2 + 1];
  int cat = ea_cat[e];
  int d = dste[e];
  float dotv = 0.f;
#pragma unroll
  for (int j = 0; j < EEMB; ++j) {
    float v = c0 * W_edge[j] + c1 * W_edge[EEMB + j] + b_edge[j] + edge_emb[cat * EEMB + j];
    eattr[(long)e * EEMB + j] = v;
    dotv += v * weae[j];
    atomicAdd(&nodesum[(long)d * EEMB + j], v);
  }
  a_e[e] = dotv;
  atomicAdd(&cnt[d], 1.f);
}

// per-node: a_s, a_d, self-loop alpha, init amax/denom
__global__ void node_prep_kernel(const float* __restrict__ hgat,
                                 const float* __restrict__ att_s,
                                 const float* __restrict__ att_d,
                                 const float* __restrict__ nodesum,
                                 const float* __restrict__ cnt,
                                 const float* __restrict__ weae,
                                 float* __restrict__ a_s, float* __restrict__ a_d,
                                 float* __restrict__ aself, float* __restrict__ amax,
                                 float* __restrict__ denom) {
  int n = blockIdx.x * blockDim.x + threadIdx.x;
  if (n >= NN) return;
  float s = 0.f, d = 0.f;
  for (int c = 0; c < CH; ++c) {
    float hv = hgat[(long)n * CH + c];
    s += hv * att_s[c];
    d += hv * att_d[c];
  }
  a_s[n] = s; a_d[n] = d;
  float cv = fmaxf(cnt[n], 1.f);
  float ae = 0.f;
#pragma unroll
  for (int j = 0; j < EEMB; ++j) ae += (nodesum[(long)n * EEMB + j] / cv) * weae[j];
  float al = leakyf_(s + d + ae);
  aself[n] = al;
  amax[n]  = al;   // self loop guarantees segment_max is well-defined
  denom[n] = 0.f;
}

__global__ void edge_alpha_kernel(const int* __restrict__ srce, const int* __restrict__ dste,
                                  const float* __restrict__ a_s, const float* __restrict__ a_d,
                                  const float* __restrict__ a_e,
                                  float* __restrict__ alpha, float* __restrict__ amax) {
  int e = blockIdx.x * blockDim.x + threadIdx.x;
  if (e >= NE) return;
  float al = leakyf_(a_s[srce[e]] + a_d[dste[e]] + a_e[e]);
  alpha[e] = al;
  atomicMaxF(&amax[dste[e]], al);
}

__global__ void edge_exp_kernel(const int* __restrict__ dste,
                                const float* __restrict__ alpha, const float* __restrict__ aself,
                                const float* __restrict__ amax,
                                float* __restrict__ exv, float* __restrict__ exself,
                                float* __restrict__ denom) {
  int i = blockIdx.x * blockDim.x + threadIdx.x;
  if (i < NE) {
    int d = dste[i];
    float ex = expf(alpha[i] - amax[d]);
    exv[i] = ex;
    atomicAdd(&denom[d], ex);
  } else if (i < NE + NN) {
    int n = i - NE;
    float ex = expf(aself[n] - amax[n]);
    exself[n] = ex;
    atomicAdd(&denom[n], ex);
  }
}

// out[dst] += w * hgat[src]  (thread per (edge, channel))
__global__ void aggregate_kernel(const int* __restrict__ srce, const int* __restrict__ dste,
                                 const float* __restrict__ exv, const float* __restrict__ denom,
                                 const float* __restrict__ hgat, float* __restrict__ outacc) {
  long idx = (long)blockIdx.x * blockDim.x + threadIdx.x;
  if (idx >= (long)NE * CH) return;
  int e = (int)(idx / CH);
  int c = (int)(idx % CH);
  int d = dste[e];
  float w = exv[e] / denom[d];
  atomicAdd(&outacc[(long)d * CH + c], w * hgat[(long)srce[e] * CH + c]);
}

// add self-loop term + bias, then BatchNorm1 (in place on outacc)
__global__ void gat_final_kernel(const float* __restrict__ hgat,
                                 const float* __restrict__ exself, const float* __restrict__ denom,
                                 const float* __restrict__ gat_b,
                                 const float* __restrict__ w, const float* __restrict__ b,
                                 const float* __restrict__ m, const float* __restrict__ v,
                                 float* __restrict__ outacc) {
  long idx = (long)blockIdx.x * blockDim.x + threadIdx.x;
  if (idx >= (long)NN * CH) return;
  int n = (int)(idx / CH);
  int c = (int)(idx % CH);
  float ws = exself[n] / denom[n];
  float val = outacc[idx] + ws * hgat[idx] + gat_b[c];
  outacc[idx] = bnf_(val, w[c], b[c], m[c], v[c]);
}

// depthwise causal conv K=4 + SiLU over xc = xz[:, :80]
__global__ void conv_silu_kernel(const float* __restrict__ xz,
                                 const float* __restrict__ cw, const float* __restrict__ cb,
                                 float* __restrict__ u) {
  long idx = (long)blockIdx.x * blockDim.x + threadIdx.x;
  if (idx >= (long)NN * CH) return;
  int n = (int)(idx / CH);
  int c = (int)(idx % CH);
  int bg = n / LG, t = n % LG;
  float acc = cb[c];
#pragma unroll
  for (int k = 0; k < 4; ++k) {
    int tt = t + k - 3;
    if (tt >= 0) acc += xz[((long)bg * LG + tt) * (2 * CH) + c] * cw[c * 4 + k];
  }
  u[idx] = siluf_(acc);
}

// proj[n, j] = sum_c u[n,c] * xp_W[c,j]   (j < 37)
__global__ void proj_kernel(const float* __restrict__ u, const float* __restrict__ xpW,
                            float* __restrict__ proj) {
  long idx = (long)blockIdx.x * blockDim.x + threadIdx.x;
  if (idx >= (long)NN * PROJW) return;
  int n = (int)(idx / PROJW);
  int j = (int)(idx % PROJW);
  float s = 0.f;
  for (int c = 0; c < CH; ++c) s += u[(long)n * CH + c] * xpW[c * PROJW + j];
  proj[idx] = s;
}

// dt[n,c] = softplus(proj[n,:5] @ dt_W + dt_b)
__global__ void dt_kernel(const float* __restrict__ proj, const float* __restrict__ dtW,
                          const float* __restrict__ dtb, float* __restrict__ dt) {
  long idx = (long)blockIdx.x * blockDim.x + threadIdx.x;
  if (idx >= (long)NN * CH) return;
  int n = (int)(idx / CH);
  int c = (int)(idx % CH);
  float pre = dtb[c];
#pragma unroll
  for (int r = 0; r < DTR; ++r) pre += proj[(long)n * PROJW + r] * dtW[r * CH + c];
  dt[idx] = (pre > 20.f) ? pre : log1pf(expf(pre));
}

// selective scan: one thread per (graph, channel); 16-state recurrence in regs.
// writes gated = y * silu(z) in place of dt buffer (safe: read-before-write per element)
__global__ void scan_kernel(const float* __restrict__ xz, const float* __restrict__ u,
                            const float* __restrict__ dt, const float* __restrict__ proj,
                            const float* __restrict__ Alog, const float* __restrict__ Dp,
                            float* __restrict__ gated) {
  int bg = blockIdx.x;
  int c  = threadIdx.x;
  if (c >= CH) return;
  float Arow[DSTATE], st[DSTATE];
#pragma unroll
  for (int s = 0; s < DSTATE; ++s) { Arow[s] = -expf(Alog[c * DSTATE + s]); st[s] = 0.f; }
  float Dv = Dp[c];
  for (int t = 0; t < LG; ++t) {
    long n = (long)bg * LG + t;
    float dtv = dt[n * CH + c];
    float uv  = u[n * CH + c];
    const float* pr = proj + n * PROJW;
    float y = 0.f;
#pragma unroll
    for (int s = 0; s < DSTATE; ++s) {
      float dA = expf(dtv * Arow[s]);
      st[s] = dA * st[s] + dtv * pr[DTR + s] * uv;       // B part
      y += st[s] * pr[DTR + DSTATE + s];                 // C part
    }
    y += uv * Dv;
    float zv = xz[n * (2 * CH) + CH + c];
    gated[n * CH + c] = y * siluf_(zv);
  }
}

// out1 = hg + bn2(hm)
__global__ void combine_kernel(const float* __restrict__ hg, const float* __restrict__ hm,
                               const float* __restrict__ w, const float* __restrict__ b,
                               const float* __restrict__ m, const float* __restrict__ v,
                               float* __restrict__ out1) {
  long idx = (long)blockIdx.x * blockDim.x + threadIdx.x;
  if (idx >= (long)NN * CH) return;
  int c = (int)(idx % CH);
  out1[idx] = hg[idx] + bnf_(hm[idx], w[c], b[c], m[c], v[c]);
}

// relu(bn3(out1 + t2)) -> mean pool via atomics
__global__ void final_node_kernel(const float* __restrict__ out1, const float* __restrict__ t2,
                                  const float* __restrict__ w, const float* __restrict__ b,
                                  const float* __restrict__ m, const float* __restrict__ v,
                                  float* __restrict__ pooled) {
  long idx = (long)blockIdx.x * blockDim.x + threadIdx.x;
  if (idx >= (long)NN * CH) return;
  int n = (int)(idx / CH);
  int c = (int)(idx % CH);
  float x = bnf_(out1[idx] + t2[idx], w[c], b[c], m[c], v[c]);
  x = fmaxf(x, 0.f);
  int g = n / LG;
  atomicAdd(&pooled[(long)g * CH + c], x * (1.f / (float)LG));
}

// tiny per-graph MLP head: pooled -> 40 -> 20 -> 1
__global__ void head_kernel(const float* __restrict__ pooled,
                            const float* __restrict__ W1, const float* __restrict__ b1,
                            const float* __restrict__ W2, const float* __restrict__ b2,
                            const float* __restrict__ W3, const float* __restrict__ b3,
                            float* __restrict__ pred) {
  __shared__ float p[CH], h1[40], h2[20];
  int g = blockIdx.x, tid = threadIdx.x;
  if (tid < CH) p[tid] = pooled[(long)g * CH + tid];
  __syncthreads();
  if (tid < 40) {
    float s = b1[tid];
    for (int k = 0; k < CH; ++k) s += p[k] * W1[k * 40 + tid];
    h1[tid] = fmaxf(s, 0.f);
  }
  __syncthreads();
  if (tid < 20) {
    float s = b2[tid];
    for (int k = 0; k < 40; ++k) s += h1[k] * W2[k * 20 + tid];
    h2[tid] = fmaxf(s, 0.f);
  }
  __syncthreads();
  if (tid == 0) {
    float s = b3[0];
    for (int k = 0; k < 20; ++k) s += h2[k] * W3[k];
    pred[g] = s;
  }
}

// ---------------------------------------------------------------------------
extern "C" void kernel_launch(void* const* d_in, const int* in_sizes, int n_in,
                              void* d_out, int out_size, void* d_ws, size_t ws_size,
                              hipStream_t stream) {
  const float* x        = (const float*)d_in[0];
  const int*   ei       = (const int*)d_in[1];
  const int*   srce     = ei;
  const int*   dste     = ei + NE;
  const int*   ea_cat   = (const int*)d_in[2];
  const float* ea_cont  = (const float*)d_in[3];
  const float* W_uni    = (const float*)d_in[4];
  const float* b_uni    = (const float*)d_in[5];
  const float* edge_emb = (const float*)d_in[6];
  const float* W_edge   = (const float*)d_in[7];
  const float* b_edge   = (const float*)d_in[8];
  const float* gat_W    = (const float*)d_in[9];
  const float* gat_as   = (const float*)d_in[10];
  const float* gat_ad   = (const float*)d_in[11];
  const float* gat_We   = (const float*)d_in[12];
  const float* gat_ae   = (const float*)d_in[13];
  const float* gat_b    = (const float*)d_in[14];
  const float* bn1w = (const float*)d_in[15], *bn1b = (const float*)d_in[16];
  const float* bn1m = (const float*)d_in[17], *bn1v = (const float*)d_in[18];
  const float* bn2w = (const float*)d_in[19], *bn2b = (const float*)d_in[20];
  const float* bn2m = (const float*)d_in[21], *bn2v = (const float*)d_in[22];
  const float* bn3w = (const float*)d_in[23], *bn3b = (const float*)d_in[24];
  const float* bn3m = (const float*)d_in[25], *bn3v = (const float*)d_in[26];
  const float* m_in_W   = (const float*)d_in[27];
  const float* m_conv_w = (const float*)d_in[28];
  const float* m_conv_b = (const float*)d_in[29];
  const float* m_xp_W   = (const float*)d_in[30];
  const float* m_dt_W   = (const float*)d_in[31];
  const float* m_dt_b   = (const float*)d_in[32];
  const float* m_Alog   = (const float*)d_in[33];
  const float* m_D      = (const float*)d_in[34];
  const float* m_out_W  = (const float*)d_in[35];
  const float* mlp_W1   = (const float*)d_in[36];
  const float* mlp_b1   = (const float*)d_in[37];
  const float* mlp_W2   = (const float*)d_in[38];
  const float* mlp_b2   = (const float*)d_in[39];
  const float* f_W1 = (const float*)d_in[40], *f_b1 = (const float*)d_in[41];
  const float* f_W2 = (const float*)d_in[42], *f_b2 = (const float*)d_in[43];
  const float* f_W3 = (const float*)d_in[44], *f_b3 = (const float*)d_in[45];

  float* ws = (float*)d_ws;
  size_t off = 0;
  auto alloc = [&](size_t n) { float* p = ws + off; off += n; return p; };

  float* h       = alloc((size_t)NN * CH);       // node features (later: out1)
  float* hgat    = alloc((size_t)NN * CH);       // h @ gat_W     (later: t2)
  float* outacc  = alloc((size_t)NN * CH);       // GAT agg -> hg after bn1
  float* xz      = alloc((size_t)NN * 2 * CH);   // mamba in_proj (later: t1)
  float* ubuf    = alloc((size_t)NN * CH);       // conv+silu u   (later: hm)
  float* dtbuf   = alloc((size_t)NN * CH);       // dt            (later: gated)
  float* proj    = alloc((size_t)NN * PROJW);
  float* eattr   = alloc((size_t)NE * EEMB);
  float* a_e     = alloc(NE);
  float* alpha   = alloc(NE);
  float* exv     = alloc(NE);
  float* nodesum = alloc((size_t)NN * EEMB);
  float* cnt     = alloc(NN);
  float* a_s     = alloc(NN);
  float* a_d     = alloc(NN);
  float* amax    = alloc(NN);
  float* denom   = alloc(NN);
  float* aself   = alloc(NN);
  float* exself  = alloc(NN);
  float* weae    = alloc(16);

  float* out1  = h;      // aliases (lifetimes are disjoint)
  float* t1    = xz;
  float* t2    = hgat;
  float* hm    = ubuf;
  float* gated = dtbuf;
  float* pred   = (float*)d_out;
  float* pooled = (float*)d_out + NG;

  const int TB = 256;

  // ---- init accumulators & output ----
  zero_kernel<<<CDIV((long)NN * EEMB, TB), TB, 0, stream>>>(nodesum, (long)NN * EEMB);
  zero_kernel<<<CDIV(NN, TB), TB, 0, stream>>>(cnt, NN);
  zero_kernel<<<CDIV(out_size, TB), TB, 0, stream>>>((float*)d_out, out_size);
  zero_kernel<<<CDIV((long)NN * CH, TB), TB, 0, stream>>>(outacc, (long)NN * CH);

  // ---- node embedding: h[:, :64] = x[:,2:1026] @ W_uni + b_uni (WMMA) ----
  gemm_bf16_wmma<<<dim3(CDIV(NN, 64), CDIV(64, 64)), TB, 0, stream>>>(
      x + 2, DUNI + 2, W_uni, 64, b_uni, h, CH, NN, 64, DUNI);
  pe_kernel<<<CDIV(NN, TB), TB, 0, stream>>>(x, h);

  // ---- edge attrs + self-loop stats ----
  weae_kernel<<<1, 32, 0, stream>>>(gat_We, gat_ae, weae);
  edge_attr_kernel<<<CDIV(NE, TB), TB, 0, stream>>>(
      ea_cat, ea_cont, W_edge, b_edge, edge_emb, weae, dste, eattr, a_e, nodesum, cnt);

  // ---- hgat = h @ gat_W (WMMA) ----
  gemm_bf16_wmma<<<dim3(CDIV(NN, 64), CDIV(CH, 64)), TB, 0, stream>>>(
      h, CH, gat_W, CH, nullptr, hgat, CH, NN, CH, CH);

  // ---- GAT attention: 3-pass segment softmax + aggregation ----
  node_prep_kernel<<<CDIV(NN, TB), TB, 0, stream>>>(
      hgat, gat_as, gat_ad, nodesum, cnt, weae, a_s, a_d, aself, amax, denom);
  edge_alpha_kernel<<<CDIV(NE, TB), TB, 0, stream>>>(srce, dste, a_s, a_d, a_e, alpha, amax);
  edge_exp_kernel<<<CDIV(NE + NN, TB), TB, 0, stream>>>(dste, alpha, aself, amax, exv, exself, denom);
  aggregate_kernel<<<CDIV((long)NE * CH, TB), TB, 0, stream>>>(srce, dste, exv, denom, hgat, outacc);
  gat_final_kernel<<<CDIV((long)NN * CH, TB), TB, 0, stream>>>(
      hgat, exself, denom, gat_b, bn1w, bn1b, bn1m, bn1v, outacc);

  // ---- Mamba ----
  gemm_bf16_wmma<<<dim3(CDIV(NN, 64), CDIV(2 * CH, 64)), TB, 0, stream>>>(
      h, CH, m_in_W, 2 * CH, nullptr, xz, 2 * CH, NN, 2 * CH, CH);
  conv_silu_kernel<<<CDIV((long)NN * CH, TB), TB, 0, stream>>>(xz, m_conv_w, m_conv_b, ubuf);
  proj_kernel<<<CDIV((long)NN * PROJW, TB), TB, 0, stream>>>(ubuf, m_xp_W, proj);
  dt_kernel<<<CDIV((long)NN * CH, TB), TB, 0, stream>>>(proj, m_dt_W, m_dt_b, dtbuf);
  scan_kernel<<<NG, 128, 0, stream>>>(xz, ubuf, dtbuf, proj, m_Alog, m_D, gated);
  gemm_bf16_wmma<<<dim3(CDIV(NN, 64), CDIV(CH, 64)), TB, 0, stream>>>(
      gated, CH, m_out_W, CH, nullptr, hm, CH, NN, CH, CH);
  combine_kernel<<<CDIV((long)NN * CH, TB), TB, 0, stream>>>(
      outacc, hm, bn2w, bn2b, bn2m, bn2v, out1);

  // ---- residual MLP (WMMA) ----
  gemm_bf16_wmma<<<dim3(CDIV(NN, 64), CDIV(2 * CH, 64)), TB, 0, stream>>>(
      out1, CH, mlp_W1, 2 * CH, mlp_b1, t1, 2 * CH, NN, 2 * CH, CH);
  relu_kernel<<<CDIV((long)NN * 2 * CH, TB), TB, 0, stream>>>(t1, (long)NN * 2 * CH);
  gemm_bf16_wmma<<<dim3(CDIV(NN, 64), CDIV(CH, 64)), TB, 0, stream>>>(
      t1, 2 * CH, mlp_W2, CH, mlp_b2, t2, CH, NN, CH, 2 * CH);

  // ---- bn3 + relu + mean-pool, then head ----
  final_node_kernel<<<CDIV((long)NN * CH, TB), TB, 0, stream>>>(
      out1, t2, bn3w, bn3b, bn3m, bn3v, pooled);
  head_kernel<<<NG, 128, 0, stream>>>(pooled, f_W1, f_b1, f_W2, f_b2, f_W3, f_b3, pred);
}
